// ResVertixRefineShapenet_19069654794320
// MI455X (gfx1250) — compile-verified, matrix operand order checked by reading, not run on hardware
//
#include <hip/hip_runtime.h>
#include <hip/hip_bf16.h>
#include <math.h>
#include <stdint.h>

// ---------------- problem constants ----------------
#define VTOT   32768      // B*VP vertices
#define NFEAT  128
#define NEDGE  196608     // 6*V
#define K0PAD  288        // 259 (NF+ND+NF) padded to multiple of 32
#define CHSUM  3840

typedef __attribute__((ext_vector_type(16))) __bf16 v16bf;
typedef __attribute__((ext_vector_type(8)))  float  v8f;

union V16U { uint4 q[2]; v16bf v; };

// fp32 -> bf16 round-to-nearest-even
__device__ __forceinline__ unsigned short f2bf(float f) {
    unsigned int u = __builtin_bit_cast(unsigned int, f);
    unsigned int r = u + 0x7FFFu + ((u >> 16) & 1u);
    return (unsigned short)(r >> 16);
}

__device__ __forceinline__ v8f wmma_bf16(v16bf a, v16bf b, v8f c) {
    return __builtin_amdgcn_wmma_f32_16x16x32_bf16(
        /*neg_a=*/false, a, /*neg_b=*/false, b,
        /*c_mod=*/(short)0, c, /*reuse_a=*/false, /*reuse_b=*/false);
}

// A 16x32 bf16 fragment from row-major LDS tile (stride 32 elems).
// lanes 0-15: row=lane, K {0..7,16..23}; lanes 16-31: same rows, K {8..15,24..31}.
// Each half-fragment is 8 contiguous bf16 = one 16B DS load.
__device__ __forceinline__ v16bf frag_a_lds(const unsigned short* tile, int lane) {
    const int row = lane & 15;
    const int hb  = (lane >> 4) << 3;   // 0 or 8
    V16U u;
    u.q[0] = *(const uint4*)&tile[row * 32 + hb];
    u.q[1] = *(const uint4*)&tile[row * 32 + 16 + hb];
    return u.v;
}

// B fragment from pre-swizzled global weights: layout [ck][nt][lane][16] bf16,
// i.e. 32 contiguous bytes per lane, 1KB per (ck,nt) tile -> coalesced b128 loads.
__device__ __forceinline__ v16bf frag_b_swz(const unsigned short* __restrict__ Bsw,
                                            int ck, int nt, int lane) {
    const unsigned short* p = Bsw + (((size_t)(ck * 8 + nt) * 32 + lane) << 4);
    V16U u;
    u.q[0] = ((const uint4*)p)[0];
    u.q[1] = ((const uint4*)p)[1];
    return u.v;
}

// ---------------- weight conversion into WMMA-fragment-swizzled bf16 ----------------
// dst element i -> (ck, nt, lane, e); source (k, col) per ISA 7.12.2 B layout.
__global__ void k_cvt_swz(const float* __restrict__ src, unsigned short* __restrict__ dst,
                          int rows, int kpad) {
    int i = blockIdx.x * blockDim.x + threadIdx.x;
    int n = kpad * 128;
    if (i >= n) return;
    int ck = i >> 12;            // 4096 elements per 32-K chunk
    int r  = i & 4095;
    int nt = r >> 9;             // 8 n-tiles of 512 elements
    int l  = (r >> 4) & 31;
    int e  = r & 15;
    int hb = (l >> 4) << 3;
    int k  = ck * 32 + ((e < 8) ? (hb + e) : (16 + hb + (e - 8)));
    int col = nt * 16 + (l & 15);
    float v = (k < rows) ? src[(size_t)k * 128 + col] : 0.0f;
    dst[i] = f2bf(v);
}

// plain fp32 -> bf16 row-major (activations)
__global__ void k_cvt(const float* __restrict__ src, unsigned short* __restrict__ dst, int n) {
    int i = blockIdx.x * blockDim.x + threadIdx.x;
    if (i < n) dst[i] = f2bf(src[i]);
}

__global__ void k_zero(float* __restrict__ p, int n) {
    int i = blockIdx.x * blockDim.x + threadIdx.x;
    if (i < n) p[i] = 0.0f;
}

__global__ void k_copy(const float* __restrict__ s, float* __restrict__ d, int n) {
    int i = blockIdx.x * blockDim.x + threadIdx.x;
    if (i < n) d[i] = s[i];
}

// ---------------- fused perceptual-pooling GEMM ----------------
// out[v,n] = sum_c aligned[v,c] * w_lin[c,n]; A tile (16x32) gathered+blended into LDS per chunk.
__global__ void k_align_gemm(const float* __restrict__ f1, const float* __restrict__ f2,
                             const float* __restrict__ f3, const float* __restrict__ f4,
                             const float* __restrict__ vpos,
                             const unsigned short* __restrict__ Bsw,  // swizzled 3840x128 bf16
                             float* __restrict__ out) {               // V x 128 fp32
    __shared__ unsigned short tileA[16 * 32];
    __shared__ int   s_off[4][16][4];
    __shared__ float s_w[4][16][4];

    const int tid = threadIdx.x;
    const int v0  = blockIdx.x * 16;
    const int b   = v0 >> 12;   // batch = v / 4096

    if (tid < 64) {
        const int lv = tid >> 4, r = tid & 15;
        const int v  = v0 + r;
        const float px = vpos[v * 3 + 0], py = vpos[v * 3 + 1], pz = vpos[v * 3 + 2];
        float hh = fminf(fmaxf(248.0f * (py / pz)    + 111.5f, 0.0f), 223.0f);
        float ww = fminf(fmaxf(248.0f * (px / (-pz)) + 111.5f, 0.0f), 223.0f);
        const int S = (lv == 0) ? 56 : (lv == 1) ? 28 : (lv == 2) ? 14 : 7;
        float x = ww / (224.0f / (float)S);
        float y = hh / (224.0f / (float)S);
        int x1 = (int)floorf(x);
        int x2 = (int)ceilf(x); if (x2 > S - 1) x2 = S - 1;
        int y1 = (int)floorf(y);
        int y2 = (int)ceilf(y); if (y2 > S - 1) y2 = S - 1;
        int xi = (int)x, yi = (int)y;
        s_off[lv][r][0] = x1 * S + y1;   // Q11
        s_off[lv][r][1] = x2 * S + y1;   // Q21
        s_off[lv][r][2] = x1 * S + y2;   // Q12
        s_off[lv][r][3] = x2 * S + y2;   // Q22
        s_w[lv][r][0] = (float)((x2 - xi) * (y2 - yi));
        s_w[lv][r][1] = (float)((xi - x1) * (y2 - yi));
        s_w[lv][r][2] = (float)((x2 - xi) * (yi - y1));
        s_w[lv][r][3] = (float)((xi - x1) * (yi - y1));
    }
    __syncthreads();

    const int lane = tid & 31;
    const int wv   = tid >> 5;           // 8 waves, each owns one 16-col tile
    v8f acc;
#pragma unroll
    for (int i = 0; i < 8; ++i) acc[i] = 0.0f;

    for (int k0 = 0; k0 < CHSUM; k0 += 32) {
        const float* fp; int C, S, st, lv;
        if (k0 < 256)       { fp = f1; C = 256;  S = 56; st = 0;    lv = 0; }
        else if (k0 < 768)  { fp = f2; C = 512;  S = 28; st = 256;  lv = 1; }
        else if (k0 < 1792) { fp = f3; C = 1024; S = 14; st = 768;  lv = 2; }
        else                { fp = f4; C = 2048; S = 7;  st = 1792; lv = 3; }
        const int SS = S * S;
        // gather 16x32 A-tile: 512 elements, 2 per thread
#pragma unroll
        for (int j = 0; j < 2; ++j) {
            const int e   = tid * 2 + j;
            const int row = e >> 5, kk = e & 31;
            const int c   = k0 + kk - st;
            const float* base = fp + (size_t)(b * C + c) * SS;
            float val = s_w[lv][row][0] * base[s_off[lv][row][0]]
                      + s_w[lv][row][1] * base[s_off[lv][row][1]]
                      + s_w[lv][row][2] * base[s_off[lv][row][2]]
                      + s_w[lv][row][3] * base[s_off[lv][row][3]];
            tileA[row * 32 + kk] = f2bf(val);
        }
        __syncthreads();
        v16bf a  = frag_a_lds(tileA, lane);
        v16bf bb = frag_b_swz(Bsw, k0 >> 5, wv, lane);
        acc = wmma_bf16(a, bb, acc);
        __syncthreads();
    }

    const int colb = lane & 15;
    const int rhb  = (lane >> 4) * 8;
#pragma unroll
    for (int r = 0; r < 8; ++r) {
        out[(size_t)(v0 + rhb + r) * 128 + wv * 16 + colb] = acc[r];
    }
}

// ---------------- feat assembly: [vf(128) | pos(3) | lin(128) | zeros] -> bf16 ----------------
__global__ void k_build_feat(const float* __restrict__ vf, const float* __restrict__ pos,
                             const float* __restrict__ lin, unsigned short* __restrict__ featA) {
    int i = blockIdx.x * blockDim.x + threadIdx.x;
    if (i >= VTOT * K0PAD) return;
    int v = i / K0PAD, c = i - v * K0PAD;
    float val;
    if (c < 128)      val = vf[(size_t)v * 128 + c];
    else if (c < 131) val = pos[(size_t)v * 3 + (c - 128)];
    else if (c < 259) val = lin[(size_t)v * 128 + (c - 131)];
    else              val = 0.0f;
    featA[i] = f2bf(val);
}

// ---------------- generic WMMA GEMM, async-to-LDS double-buffered A ----------------
// out = (relu ? max(0, A@B + D) : A@B + D) + S        (D, S optional)
__global__ void k_gemm(const unsigned short* __restrict__ A,    // M x K bf16 row-major
                       const unsigned short* __restrict__ Bsw,  // K x 128 bf16, frag-swizzled
                       const float* __restrict__ Dadd,
                       const float* __restrict__ Sadd,
                       float* __restrict__ out, int K, int relu) {
    __shared__ unsigned short tileA[2][128 * 32];
    const int tid  = threadIdx.x;
    const int row0 = blockIdx.x * 128;
    const int lane = tid & 31;
    const int wv   = tid >> 5;
    const int nk   = K >> 5;

    v8f acc[8];
#pragma unroll
    for (int nt = 0; nt < 8; ++nt)
#pragma unroll
        for (int i = 0; i < 8; ++i) acc[nt][i] = 0.0f;

    const int ar = tid >> 1;            // A: 128 rows, 2 threads/row, 32B each
    const int ak = (tid & 1) * 16;

    // CDNA5 async copy: global -> LDS, tracked by ASYNCcnt (ISA 15.18.3 op 98)
    auto issue = [&](int ck, int buf) {
        unsigned lds = (unsigned)(uintptr_t)&tileA[buf][ar * 32 + ak];   // AS3 offset in low bits
        const unsigned short* ga = A + (size_t)(row0 + ar) * K + ck * 32 + ak;
        asm volatile(
            "global_load_async_to_lds_b128 %0, %1, off\n\t"
            "global_load_async_to_lds_b128 %0, %1, off offset:16"
            :: "v"(lds), "v"(ga) : "memory");
    };

    issue(0, 0);
    for (int ck = 0; ck < nk; ++ck) {
        if (ck + 1 < nk) {
            issue(ck + 1, (ck + 1) & 1);
            asm volatile("s_wait_asynccnt 0x2" ::: "memory");   // tile ck landed
        } else {
            asm volatile("s_wait_asynccnt 0x0" ::: "memory");
        }
        __syncthreads();
        v16bf a = frag_a_lds(&tileA[ck & 1][(wv * 16) * 32], lane);
#pragma unroll
        for (int nt = 0; nt < 8; ++nt) {
            acc[nt] = wmma_bf16(a, frag_b_swz(Bsw, ck, nt, lane), acc[nt]);
        }
        __syncthreads();   // protect buffer ck&1 before it is refilled
    }

    const int colb = lane & 15;
    const int rhb  = (lane >> 4) * 8;
#pragma unroll
    for (int nt = 0; nt < 8; ++nt) {
#pragma unroll
        for (int r = 0; r < 8; ++r) {
            size_t idx = (size_t)(row0 + wv * 16 + rhb + r) * 128 + nt * 16 + colb;
            float val = acc[nt][r];
            if (Dadd) val += Dadd[idx];
            if (relu) val = fmaxf(val, 0.0f);
            if (Sadd) val += Sadd[idx];
            out[idx] = val;
        }
    }
}

// ---------------- edge scatter-sum: agg[dst] += t[src]  (F feats) ----------------
__global__ void k_scatter(const float* __restrict__ t, const int* __restrict__ src,
                          const int* __restrict__ dst, float* __restrict__ agg, int F) {
    int i = blockIdx.x * blockDim.x + threadIdx.x;
    if (i >= NEDGE * F) return;
    int e = i / F, f = i - e * F;
    atomicAdd(&agg[(size_t)dst[e] * F + f], t[(size_t)src[e] * F + f]);
}

// ---------------- small ND=3 ops ----------------
__global__ void k_matvec3(const float* __restrict__ h, const float* __restrict__ w /*128x3*/,
                          float* __restrict__ out3) {
    int i = blockIdx.x * blockDim.x + threadIdx.x;
    if (i >= VTOT * 3) return;
    int v = i / 3, d = i - v * 3;
    float s = 0.0f;
#pragma unroll 8
    for (int k = 0; k < 128; ++k) s += h[(size_t)v * 128 + k] * w[k * 3 + d];
    out3[i] = s;
}

__global__ void k_final_pos(const float* __restrict__ h, const float* __restrict__ gw0,
                            const float* __restrict__ agg3, const float* __restrict__ pos,
                            float* __restrict__ outpos) {
    int i = blockIdx.x * blockDim.x + threadIdx.x;
    if (i >= VTOT * 3) return;
    int v = i / 3, d = i - v * 3;
    float s = 0.0f;
#pragma unroll 8
    for (int k = 0; k < 128; ++k) s += h[(size_t)v * 128 + k] * gw0[k * 3 + d];
    s += agg3[i];
    s = fmaxf(s, 0.0f);
    outpos[i] = pos[i] + tanhf(s);
}

// ---------------- host orchestration ----------------
extern "C" void kernel_launch(void* const* d_in, const int* in_sizes, int n_in,
                              void* d_out, int out_size, void* d_ws, size_t ws_size,
                              hipStream_t stream) {
    const float* f1   = (const float*)d_in[0];
    const float* f2   = (const float*)d_in[1];
    const float* f3   = (const float*)d_in[2];
    const float* f4   = (const float*)d_in[3];
    const float* vpos = (const float*)d_in[4];
    const float* vfeat= (const float*)d_in[5];
    const int*   eidx = (const int*)d_in[6];
    const float* w_lin= (const float*)d_in[7];
    const float* wr[13]; // r0p r0a0 r0a1 r0b0 r0b1 r1a0 r1a1 r1b0 r1b1 r2a0 r2a1 r2b0 r2b1
    for (int i = 0; i < 13; ++i) wr[i] = (const float*)d_in[8 + i];
    const float* gw0 = (const float*)d_in[21];
    const float* gw1 = (const float*)d_in[22];
    const int* esrc = eidx;
    const int* edst = eidx + NEDGE;

    // ---- workspace layout ----
    size_t off = 0;
    auto alloc = [&](size_t bytes) -> char* {
        size_t o = (off + 255) & ~(size_t)255;
        off = o + bytes;
        return (char*)d_ws + o;
    };
    unsigned short* w_lin_bf = (unsigned short*)alloc((size_t)CHSUM * 128 * 2);
    unsigned short* r0p_bf   = (unsigned short*)alloc((size_t)K0PAD * 128 * 2);
    unsigned short* r0a0_bf  = (unsigned short*)alloc((size_t)K0PAD * 128 * 2);
    unsigned short* r0a1_bf  = (unsigned short*)alloc((size_t)K0PAD * 128 * 2);
    unsigned short* wsq[10];
    for (int i = 0; i < 10; ++i) wsq[i] = (unsigned short*)alloc((size_t)128 * 128 * 2);
    unsigned short* featA = (unsigned short*)alloc((size_t)VTOT * K0PAD * 2);
    unsigned short* bfX   = (unsigned short*)alloc((size_t)VTOT * 128 * 2);
    unsigned short* bfG   = (unsigned short*)alloc((size_t)VTOT * 128 * 2);
    float* F0 = (float*)alloc((size_t)VTOT * 128 * 4);   // lin / t
    float* F1 = (float*)alloc((size_t)VTOT * 128 * 4);   // agg
    float* F2 = (float*)alloc((size_t)VTOT * 128 * 4);   // g1
    float* F3 = (float*)alloc((size_t)VTOT * 128 * 4);   // skip / h
    float* F4 = (float*)alloc((size_t)VTOT * 128 * 4);   // h
    float* t3   = (float*)alloc((size_t)VTOT * 3 * 4);
    float* agg3 = (float*)alloc((size_t)VTOT * 3 * 4);
    (void)ws_size; (void)in_sizes; (void)n_in; (void)out_size;

    auto cdiv = [](long a, long b) { return (int)((a + b - 1) / b); };

    // ---- weights -> bf16, swizzled into WMMA fragment order ----
    k_cvt_swz<<<cdiv((long)CHSUM * 128, 256), 256, 0, stream>>>(w_lin, w_lin_bf, CHSUM, CHSUM);
    k_cvt_swz<<<cdiv((long)K0PAD * 128, 256), 256, 0, stream>>>(wr[0], r0p_bf, 259, K0PAD);
    k_cvt_swz<<<cdiv((long)K0PAD * 128, 256), 256, 0, stream>>>(wr[1], r0a0_bf, 259, K0PAD);
    k_cvt_swz<<<cdiv((long)K0PAD * 128, 256), 256, 0, stream>>>(wr[2], r0a1_bf, 259, K0PAD);
    for (int i = 0; i < 10; ++i)
        k_cvt_swz<<<cdiv(128L * 128, 256), 256, 0, stream>>>(wr[3 + i], wsq[i], 128, 128);

    // ---- fused pooling + aligned@w_lin ----
    k_align_gemm<<<VTOT / 16, 256, 0, stream>>>(f1, f2, f3, f4, vpos, w_lin_bf, F0);
    k_build_feat<<<cdiv((long)VTOT * K0PAD, 256), 256, 0, stream>>>(vfeat, vpos, F0, featA);

    auto gemm = [&](const unsigned short* A, int K, const unsigned short* B,
                    const float* D, const float* S, int relu, float* out) {
        k_gemm<<<VTOT / 128, 256, 0, stream>>>(A, B, D, S, out, K, relu);
    };
    auto scatter = [&](const float* t, float* agg) {
        k_zero<<<cdiv((long)VTOT * 128, 256), 256, 0, stream>>>(agg, VTOT * 128);
        k_scatter<<<cdiv((long)NEDGE * 128, 256), 256, 0, stream>>>(t, esrc, edst, agg, 128);
    };
    auto cvt = [&](const float* s, unsigned short* d) {
        k_cvt<<<cdiv((long)VTOT * 128, 256), 256, 0, stream>>>(s, d, VTOT * 128);
    };

    // ---- resg block 0 (with projection skip) ----
    gemm(featA, K0PAD, r0a1_bf, nullptr, nullptr, 0, F0);        // t = feat @ a1
    scatter(F0, F1);                                             // agg
    gemm(featA, K0PAD, r0a0_bf, F1, nullptr, 1, F2);             // g1 = relu(feat@a0 + agg)
    cvt(F2, bfG);
    gemm(bfG, 128, wsq[1] /*r0b1*/, nullptr, nullptr, 0, F0);
    scatter(F0, F1);
    gemm(featA, K0PAD, r0p_bf, nullptr, nullptr, 0, F3);         // skip = feat @ r0p
    gemm(bfG, 128, wsq[0] /*r0b0*/, F1, F3, 1, F4);              // h = relu(g1@b0+agg) + skip
    cvt(F4, bfX);

    // ---- resg block 1 (identity skip = F4) ----
    gemm(bfX, 128, wsq[3] /*r1a1*/, nullptr, nullptr, 0, F0);
    scatter(F0, F1);
    gemm(bfX, 128, wsq[2] /*r1a0*/, F1, nullptr, 1, F2);
    cvt(F2, bfG);
    gemm(bfG, 128, wsq[5] /*r1b1*/, nullptr, nullptr, 0, F0);
    scatter(F0, F1);
    gemm(bfG, 128, wsq[4] /*r1b0*/, F1, F4, 1, F3);              // h -> F3
    cvt(F3, bfX);

    // ---- resg block 2 (identity skip = F3) ----
    gemm(bfX, 128, wsq[7] /*r2a1*/, nullptr, nullptr, 0, F0);
    scatter(F0, F1);
    gemm(bfX, 128, wsq[6] /*r2a0*/, F1, nullptr, 1, F2);
    cvt(F2, bfG);
    gemm(bfG, 128, wsq[9] /*r2b1*/, nullptr, nullptr, 0, F0);
    scatter(F0, F1);
    gemm(bfG, 128, wsq[8] /*r2b0*/, F1, F3, 1, F4);              // final h -> F4

    // ---- final position update (ND=3, VALU) ----
    k_matvec3<<<cdiv((long)VTOT * 3, 256), 256, 0, stream>>>(F4, gw1, t3);
    k_zero<<<cdiv((long)VTOT * 3, 256), 256, 0, stream>>>(agg3, VTOT * 3);
    k_scatter<<<cdiv((long)NEDGE * 3, 256), 256, 0, stream>>>(t3, esrc, edst, agg3, 3);
    float* outpos = (float*)d_out;
    k_final_pos<<<cdiv((long)VTOT * 3, 256), 256, 0, stream>>>(F4, gw0, agg3, vpos, outpos);
    k_copy<<<cdiv((long)VTOT * 128, 256), 256, 0, stream>>>(F4, outpos + (size_t)VTOT * 3, VTOT * 128);
}
